// MHs_88338887344453
// MI455X (gfx1250) — matrix operand order repaired and checked by reading, not compiled, not run on hardware
//
#include <hip/hip_runtime.h>
#include <hip/hip_bf16.h>
#include <math.h>

#define QDIM   8
#define MDIM   65536
#define DDIM   64
#define BDIM   128
#define UDIM   64
#define DELTA  16
#define GAMMA  0.1f

#define SUB    64                 // key rows per sub-iteration
#define NSUB   16                 // sub-iterations per chunk
#define CHUNK  (SUB * NSUB)       // 1024 keys per workgroup
#define NCHUNK (MDIM / CHUNK)     // 64 chunks
#define KPITCH 72                 // bf16 pitch for LDS key tile rows (16B aligned)
#define RPITCH 68                 // f32 pitch for LDS response rows

typedef __attribute__((ext_vector_type(16))) __bf16 v16bf;
typedef __attribute__((ext_vector_type(8)))  float  v8f;

union Frag { v16bf v; uint4 u[2]; };

// ---------------- Kernel 1: normalize x, convert to bf16 ----------------
__global__ __launch_bounds__(128) void prep_x(const float* __restrict__ x,
                                              __bf16* __restrict__ xbf) {
  const int b = threadIdx.x;
  const float* xr = x + b * DDIM;
  float ss = 0.f;
#pragma unroll
  for (int k = 0; k < DDIM; ++k) { float v = xr[k]; ss += v * v; }
  const float inv = 1.0f / fmaxf(sqrtf(ss), 1e-12f);
#pragma unroll
  for (int k = 0; k < DDIM; ++k) xbf[b * DDIM + k] = (__bf16)(xr[k] * inv);
}

// ------- Kernel 2: bf16 WMMA GEMM (x_n · K / (||K||*sqrt(d))) + per-chunk top-16 -------
__global__ __launch_bounds__(256) void gemm_topk(const float* __restrict__ Kmat,
                                                 const __bf16* __restrict__ xbf,
                                                 float* __restrict__ cand_val,
                                                 int* __restrict__ cand_idx) {
  __shared__ __bf16 ktile[SUB * KPITCH];     // 9216 B
  __shared__ float  resp[BDIM * RPITCH];     // 34816 B
  __shared__ float  lscale[SUB];             // norm accumulator -> scale

  const int chunk  = blockIdx.x;
  const int q      = blockIdx.y;
  const int tid    = threadIdx.x;
  const int lane   = tid & 31;
  const int wave   = tid >> 5;
  const int lane16 = lane & 15;
  const int laneHi = lane >> 4;

  // A fragments: this wave owns b rows [wave*16, wave*16+16). Constant across subs.
  // 16-bit A 16x32 layout: lanes<16 hold K{0..7,16..23}, lanes>=16 hold K{8..15,24..31}.
  Frag a0, a1;
  {
    const __bf16* xr = xbf + (size_t)(wave * 16 + lane16) * DDIM;
    const int b0 = laneHi * 8;
    a0.u[0] = *(const uint4*)(xr + b0);
    a0.u[1] = *(const uint4*)(xr + b0 + 16);
    a1.u[0] = *(const uint4*)(xr + 32 + b0);
    a1.u[1] = *(const uint4*)(xr + 32 + b0 + 16);
  }

  float topv[DELTA];
  int   topi[DELTA];
#pragma unroll
  for (int k = 0; k < DELTA; ++k) { topv[k] = -3.4e38f; topi[k] = 0; }

  for (int s = 0; s < NSUB; ++s) {
    const int m0 = chunk * CHUNK + s * SUB;

    if (tid < SUB) lscale[tid] = 0.f;
    __syncthreads();

    // Stage K tile to LDS as bf16; accumulate row sum-of-squares via DS atomics.
    {
      const float4* Kb = (const float4*)(Kmat + ((size_t)q * MDIM + m0) * DDIM);
#pragma unroll
      for (int it = 0; it < (SUB * 16) / 256; ++it) {
        const int i   = tid + it * 256;
        const float4 v = Kb[i];
        const int row = i >> 4;
        const int kq  = (i & 15) << 2;
        atomicAdd(&lscale[row], v.x * v.x + v.y * v.y + v.z * v.z + v.w * v.w);
        __bf16* d = &ktile[row * KPITCH + kq];
        d[0] = (__bf16)v.x; d[1] = (__bf16)v.y; d[2] = (__bf16)v.z; d[3] = (__bf16)v.w;
      }
    }
    __syncthreads();
    if (tid < SUB) lscale[tid] = 1.0f / (fmaxf(sqrtf(lscale[tid]), 1e-12f) * 8.0f);
    __syncthreads();

    // GEMM: each wave does 4 n-tiles of 16, K=64 as two 16x16x32 bf16 WMMAs.
#pragma unroll
    for (int t = 0; t < SUB / 16; ++t) {
      // B 32x16 layout: lanes<16 hold K{0..7,16..23} of column N=lane, lanes>=16 hold K{8..15,24..31}.
      Frag bb0, bb1;
      const __bf16* kr = &ktile[(t * 16 + lane16) * KPITCH];
      const int kb = laneHi * 8;
      bb0.u[0] = *(const uint4*)(kr + kb);
      bb0.u[1] = *(const uint4*)(kr + kb + 16);
      bb1.u[0] = *(const uint4*)(kr + 32 + kb);
      bb1.u[1] = *(const uint4*)(kr + 32 + kb + 16);

      v8f acc = {};
      acc = __builtin_amdgcn_wmma_f32_16x16x32_bf16(false, a0.v, false, bb0.v,
                                                    (short)0, acc, false, false);
      acc = __builtin_amdgcn_wmma_f32_16x16x32_bf16(false, a1.v, false, bb1.v,
                                                    (short)0, acc, false, false);

      const int col  = t * 16 + lane16;            // D: N = lane%16 for both halves
      const float sc = lscale[col];
      const int rbase = wave * 16 + laneHi * 8;    // D: M = r (+8 for hi lanes)
#pragma unroll
      for (int r = 0; r < 8; ++r)
        resp[(rbase + r) * RPITCH + col] = acc[r] * sc;
    }
    __syncthreads();

    // Register-resident top-16 per b row (threads 0..127), fully unrolled insertion.
    if (tid < BDIM) {
#pragma unroll 1
      for (int c = 0; c < SUB; ++c) {
        const float v = resp[tid * RPITCH + c];
        if (v > topv[DELTA - 1]) {
          const int idx = m0 + c;
#pragma unroll
          for (int k = DELTA - 1; k >= 0; --k) {
            const bool fromPrev = (k > 0) && (v > topv[k - 1]);
            const float pv = (k > 0) ? topv[k - 1] : v;
            const int   pi = (k > 0) ? topi[k - 1] : idx;
            if (v > topv[k]) {
              topv[k] = fromPrev ? pv : v;
              topi[k] = fromPrev ? pi : idx;
            }
          }
        }
      }
    }
    __syncthreads();
  }

  if (tid < BDIM) {
    const size_t base = ((size_t)(tid * QDIM + q) * NCHUNK + chunk) * DELTA;
#pragma unroll
    for (int k = 0; k < DELTA; ++k) {
      cand_val[base + k] = topv[k];
      cand_idx[base + k] = topi[k];
    }
  }
}

// ---- Kernel 3: merge candidates -> exact top-16, softmax, gather-weighted sum of M ----
__global__ __launch_bounds__(64) void merge_out(const float* __restrict__ cand_val,
                                                const int* __restrict__ cand_idx,
                                                const float* __restrict__ Mmat,
                                                float* __restrict__ out) {
  __shared__ float sval[NCHUNK * DELTA];
  __shared__ int   sidx[NCHUNK * DELTA];
  __shared__ float salpha[DELTA];
  __shared__ int   stopi[DELTA];

  const int bq  = blockIdx.x;            // b*8 + q
  const int q   = bq & (QDIM - 1);
  const int tid = threadIdx.x;

  const size_t base = (size_t)bq * (NCHUNK * DELTA);
  for (int i = tid; i < NCHUNK * DELTA; i += 64) {
    sval[i] = cand_val[base + i];
    sidx[i] = cand_idx[base + i];
  }
  __syncthreads();

  if (tid == 0) {
    float topv[DELTA]; int topi[DELTA];
#pragma unroll
    for (int k = 0; k < DELTA; ++k) { topv[k] = -3.4e38f; topi[k] = 0; }
#pragma unroll 1
    for (int i = 0; i < NCHUNK * DELTA; ++i) {
      const float v = sval[i];
      if (v > topv[DELTA - 1]) {
        const int idx = sidx[i];
#pragma unroll
        for (int k = DELTA - 1; k >= 0; --k) {
          const bool fromPrev = (k > 0) && (v > topv[k - 1]);
          const float pv = (k > 0) ? topv[k - 1] : v;
          const int   pi = (k > 0) ? topi[k - 1] : idx;
          if (v > topv[k]) {
            topv[k] = fromPrev ? pv : v;
            topi[k] = fromPrev ? pi : idx;
          }
        }
      }
    }
    const float mx = topv[0];
    float e[DELTA];
    float ssum = 0.f;
#pragma unroll
    for (int k = 0; k < DELTA; ++k) { e[k] = __expf(GAMMA * (topv[k] - mx)); ssum += e[k]; }
    const float rinv = 1.0f / ssum;
#pragma unroll
    for (int k = 0; k < DELTA; ++k) { salpha[k] = e[k] * rinv; stopi[k] = topi[k]; }
  }
  __syncthreads();

  const float* Mq = Mmat + (size_t)q * MDIM * UDIM;
  float w = 0.f;
#pragma unroll
  for (int k = 0; k < DELTA; ++k)
    w += salpha[k] * Mq[(size_t)stopi[k] * UDIM + tid];
  out[(size_t)bq * UDIM + tid] = w;
}

extern "C" void kernel_launch(void* const* d_in, const int* in_sizes, int n_in,
                              void* d_out, int out_size, void* d_ws, size_t ws_size,
                              hipStream_t stream) {
  const float* x = (const float*)d_in[0];   // [128, 64]
  const float* K = (const float*)d_in[1];   // [8, 65536, 64]
  const float* M = (const float*)d_in[2];   // [8, 65536, 64]
  float* out = (float*)d_out;               // [128, 8, 64]

  char* ws = (char*)d_ws;
  __bf16* xbf = (__bf16*)ws;                                         // 16 KB
  const size_t cand_elems = (size_t)BDIM * QDIM * NCHUNK * DELTA;    // 1M
  float* cand_val = (float*)(ws + 16 * 1024);                        // 4 MB
  int*   cand_idx = (int*)(ws + 16 * 1024 + cand_elems * sizeof(float)); // 4 MB

  prep_x<<<1, 128, 0, stream>>>(x, xbf);
  dim3 grid(NCHUNK, QDIM);
  gemm_topk<<<grid, 256, 0, stream>>>(K, xbf, cand_val, cand_idx);
  merge_out<<<BDIM * QDIM, 64, 0, stream>>>(cand_val, cand_idx, M, out);
}
